// RowSelfAttention_21002390077475
// MI455X (gfx1250) — matrix operand order repaired and checked by reading, not compiled, not run on hardware
//
#include <hip/hip_runtime.h>
#include <math.h>

// ---------------- problem constants ----------------
#define R_   128
#define C_   256
#define E_   768
#define H_   12
#define DK_  64
#define M_   (R_ * C_)        // 32768 flattened rows (r*C + i), batch == 1

typedef __attribute__((ext_vector_type(16))) __bf16 v16bf;
typedef __attribute__((ext_vector_type(8)))  float  v8f;

// ---------------- WMMA core ----------------
__device__ __forceinline__ v8f wmma_bf16(v16bf a, v16bf b, v8f c) {
  // (neg_a, A, neg_b, B, c_mod, C, reuse_a, reuse_b)
  return __builtin_amdgcn_wmma_f32_16x16x32_bf16(false, a, false, b, (short)0, c,
                                                 false, false);
}

// A fragment: row-major source, 16x32 tile at (row0, k0), contiguous in k.
// Layout (ISA 7.12.2, 16-bit A 16x32): lanes 0-15 -> M=lane, K = {0..7, 16..23};
// lanes 16-31 -> M=lane-16, K = {8..15, 24..31}.
__device__ __forceinline__ v16bf load_a_rowmajor(const __bf16* __restrict__ A,
                                                 int lda, int row0, int k0) {
  int lane = threadIdx.x & 31;
  int m  = lane & 15;
  int kb = (lane >> 4) << 3;              // 0 or 8
  const __bf16* p = A + (size_t)(row0 + m) * lda + (k0 + kb);
  union { v16bf v; uint4 u[2]; } f;
  f.u[0] = *(const uint4*)(p);            // K = kb+0 .. kb+7
  f.u[1] = *(const uint4*)(p + 16);       // K = 16+kb .. 16+kb+7
  return f.v;
}

// B fragment from W^T: B[k][n] = W[n][k], W row-major [N, ldw], contiguous in k.
// Layout: lanes 0-15 -> N=lane, K = {0..7, 16..23}; lanes 16-31 -> N=lane-16, K = {8..15, 24..31}.
__device__ __forceinline__ v16bf load_b_wt(const __bf16* __restrict__ W,
                                           int ldw, int n0, int k0) {
  int lane = threadIdx.x & 31;
  int n  = lane & 15;
  int kb = (lane >> 4) << 3;              // 0 or 8
  const __bf16* p = W + (size_t)(n0 + n) * ldw + (k0 + kb);
  union { v16bf v; uint4 u[2]; } f;
  f.u[0] = *(const uint4*)(p);
  f.u[1] = *(const uint4*)(p + 16);
  return f.v;
}

// C/D 16x16 f32 layout: VGPR j -> M = j + 8*(lane>=16), N = lane&15.
__device__ __forceinline__ void store_tile_f32(float* __restrict__ Cp, int ldc,
                                               int row0, int n0, v8f acc,
                                               const float* __restrict__ bias) {
  int lane = threadIdx.x & 31;
  int n  = lane & 15;
  int mh = (lane >> 4) << 3;
  float badd = bias ? bias[n0 + n] : 0.0f;
#pragma unroll
  for (int j = 0; j < 8; ++j) {
    Cp[(size_t)(row0 + mh + j) * ldc + (n0 + n)] = acc[j] + badd;
  }
}

__device__ __forceinline__ void store_tile_bf16(__bf16* __restrict__ Cp, int ldc,
                                                int row0, int n0, v8f acc,
                                                const float* __restrict__ bias,
                                                float scale) {
  int lane = threadIdx.x & 31;
  int n  = lane & 15;
  int mh = (lane >> 4) << 3;
  float badd = bias ? bias[n0 + n] : 0.0f;
#pragma unroll
  for (int j = 0; j < 8; ++j) {
    Cp[(size_t)(row0 + mh + j) * ldc + (n0 + n)] = (__bf16)((acc[j] + badd) * scale);
  }
}

// ---------------- kernels ----------------

// fp32 -> bf16 convert, 4-wide (n must be a multiple of 4)
__global__ void k_cvt_bf16(const float* __restrict__ src, __bf16* __restrict__ dst,
                           int n) {
  int i = blockIdx.x * blockDim.x + threadIdx.x;
  int stride = gridDim.x * blockDim.x;
  int n4 = n >> 2;
  for (; i < n4; i += stride) {
    float4 f = ((const float4*)src)[i];
    union { __bf16 h[4]; uint2 u; } o;
    o.h[0] = (__bf16)f.x; o.h[1] = (__bf16)f.y;
    o.h[2] = (__bf16)f.z; o.h[3] = (__bf16)f.w;
    ((uint2*)dst)[i] = o.u;
  }
}

// v[m=r*C+j][h*64+d] -> vt[h][r][d][j]  (makes ctx B fragments contiguous)
__global__ void k_transpose_v(const __bf16* __restrict__ vb,
                              __bf16* __restrict__ vt) {
  int i = blockIdx.x * blockDim.x + threadIdx.x;
  int stride = gridDim.x * blockDim.x;
  const int total = H_ * R_ * DK_ * C_;        // 25,165,824
  for (; i < total; i += stride) {
    int j = i & (C_ - 1);
    int d = (i >> 8) & (DK_ - 1);
    int r = (i >> 14) & (R_ - 1);
    int h = i >> 21;
    vt[i] = vb[(size_t)(r * C_ + j) * E_ + h * DK_ + d];
  }
}

// out_bf16[M_ x E_] = (X @ W^T + bias) * scale, 64x32 register-blocked (4x2 frags)
// waves = (M_/64)*(E_/32) = 512*24 = 12288
__global__ void __launch_bounds__(128, 1)
k_proj_gemm(const __bf16* __restrict__ X, const __bf16* __restrict__ W,
            const float* __restrict__ bias, __bf16* __restrict__ out, float scale) {
  int wave = blockIdx.x * (blockDim.x >> 5) + (threadIdx.x >> 5);
  const int ntn = E_ / 32;                 // 24
  int mt = wave / ntn, nt = wave - mt * ntn;
  int row0 = mt << 6, n0 = nt << 5;
  v8f acc[4][2];
#pragma unroll
  for (int i = 0; i < 4; ++i)
#pragma unroll
    for (int j = 0; j < 2; ++j) acc[i][j] = (v8f){0.f,0.f,0.f,0.f,0.f,0.f,0.f,0.f};
  for (int k0 = 0; k0 < E_; k0 += 32) {
    v16bf a[4], b[2];
#pragma unroll
    for (int i = 0; i < 4; ++i) a[i] = load_a_rowmajor(X, E_, row0 + 16 * i, k0);
#pragma unroll
    for (int j = 0; j < 2; ++j) b[j] = load_b_wt(W, E_, n0 + 16 * j, k0);
#pragma unroll
    for (int i = 0; i < 4; ++i)
#pragma unroll
      for (int j = 0; j < 2; ++j) acc[i][j] = wmma_bf16(a[i], b[j], acc[i][j]);
  }
#pragma unroll
  for (int i = 0; i < 4; ++i)
#pragma unroll
    for (int j = 0; j < 2; ++j)
      store_tile_bf16(out, E_, row0 + 16 * i, n0 + 16 * j, acc[i][j], bias, scale);
}

// same GEMM, fp32 output into d_out (final projection), 64x32 blocked
__global__ void __launch_bounds__(128, 1)
k_out_gemm(const __bf16* __restrict__ X, const __bf16* __restrict__ W,
           const float* __restrict__ bias, float* __restrict__ out) {
  int wave = blockIdx.x * (blockDim.x >> 5) + (threadIdx.x >> 5);
  const int ntn = E_ / 32;
  int mt = wave / ntn, nt = wave - mt * ntn;
  int row0 = mt << 6, n0 = nt << 5;
  v8f acc[4][2];
#pragma unroll
  for (int i = 0; i < 4; ++i)
#pragma unroll
    for (int j = 0; j < 2; ++j) acc[i][j] = (v8f){0.f,0.f,0.f,0.f,0.f,0.f,0.f,0.f};
  for (int k0 = 0; k0 < E_; k0 += 32) {
    v16bf a[4], b[2];
#pragma unroll
    for (int i = 0; i < 4; ++i) a[i] = load_a_rowmajor(X, E_, row0 + 16 * i, k0);
#pragma unroll
    for (int j = 0; j < 2; ++j) b[j] = load_b_wt(W, E_, n0 + 16 * j, k0);
#pragma unroll
    for (int i = 0; i < 4; ++i)
#pragma unroll
      for (int j = 0; j < 2; ++j) acc[i][j] = wmma_bf16(a[i], b[j], acc[i][j]);
  }
#pragma unroll
  for (int i = 0; i < 4; ++i)
#pragma unroll
    for (int j = 0; j < 2; ++j)
      store_tile_f32(out, E_, row0 + 16 * i, n0 + 16 * j, acc[i][j], bias);
}

// attn[h][i][j] = sum_{r,d} q[r,i,h,d] * k[r,j,h,d]; 32x32 blocked (2x2 frags)
// waves = H_ * (C_/32)^2 = 12 * 64 = 768 ; K = R_*DK_ = 8192
__global__ void __launch_bounds__(128, 1)
k_attn_gemm(const __bf16* __restrict__ qb, const __bf16* __restrict__ kb,
            float* __restrict__ attn) {
  int wave = blockIdx.x * (blockDim.x >> 5) + (threadIdx.x >> 5);
  int h = wave >> 6;                       // 64 tiles per head
  int t = wave & 63;
  int i0 = (t >> 3) << 5;
  int j0 = (t & 7) << 5;
  v8f acc[2][2];
#pragma unroll
  for (int i = 0; i < 2; ++i)
#pragma unroll
    for (int j = 0; j < 2; ++j) acc[i][j] = (v8f){0.f,0.f,0.f,0.f,0.f,0.f,0.f,0.f};
  for (int k0 = 0; k0 < R_ * DK_; k0 += 32) {
    int r  = k0 >> 6;                      // DK_ = 64: each 32-chunk sits in one r
    int d0 = k0 & 63;
    const __bf16* qbase = qb + (size_t)r * C_ * E_ + h * DK_ + d0;
    const __bf16* kbase = kb + (size_t)r * C_ * E_ + h * DK_ + d0;
    v16bf a[2], b[2];
#pragma unroll
    for (int i = 0; i < 2; ++i) a[i] = load_a_rowmajor(qbase, E_, i0 + 16 * i, 0);
#pragma unroll
    for (int j = 0; j < 2; ++j) b[j] = load_b_wt(kbase, E_, j0 + 16 * j, 0);
#pragma unroll
    for (int i = 0; i < 2; ++i)
#pragma unroll
      for (int j = 0; j < 2; ++j) acc[i][j] = wmma_bf16(a[i], b[j], acc[i][j]);
  }
  float* Ah = attn + (size_t)h * C_ * C_;
#pragma unroll
  for (int i = 0; i < 2; ++i)
#pragma unroll
    for (int j = 0; j < 2; ++j)
      store_tile_f32(Ah, C_, i0 + 16 * i, j0 + 16 * j, acc[i][j], nullptr);
}

// row softmax over j; fp32 probs to d_out region + bf16 copy for context GEMM
__global__ void k_softmax(const float* __restrict__ attn,
                          float* __restrict__ probs,
                          __bf16* __restrict__ probsb) {
  __shared__ float red[256];
  int row = blockIdx.x;                    // [0, H_*C_)
  int tid = threadIdx.x;                   // 256 threads
  float v = attn[(size_t)row * C_ + tid];
  red[tid] = v;
  __syncthreads();
  for (int s = 128; s > 0; s >>= 1) {
    if (tid < s) red[tid] = fmaxf(red[tid], red[tid + s]);
    __syncthreads();
  }
  float mx = red[0];
  __syncthreads();
  float e = __expf(v - mx);
  red[tid] = e;
  __syncthreads();
  for (int s = 128; s > 0; s >>= 1) {
    if (tid < s) red[tid] += red[tid + s];
    __syncthreads();
  }
  float p = e / red[0];
  probs[(size_t)row * C_ + tid]  = p;
  probsb[(size_t)row * C_ + tid] = (__bf16)p;
}

// c[r,i,h,d] = sum_j probs[h,i,j] * vt[h,r,d,j]; 64(i) x 32(d) blocked (4x2)
// waves = H_ * R_ * (C_/64) * (DK_/32) = 12*128*4*2 = 12288 ; K = C_ = 256
__global__ void __launch_bounds__(128, 1)
k_ctx_gemm(const __bf16* __restrict__ probsb, const __bf16* __restrict__ vt,
           __bf16* __restrict__ cb) {
  int wave = blockIdx.x * (blockDim.x >> 5) + (threadIdx.x >> 5);
  int db = wave & 1;         wave >>= 1;
  int ib = wave & 3;         wave >>= 2;
  int r  = wave & (R_ - 1);  wave >>= 7;
  int h  = wave;
  int i0 = ib << 6, d0 = db << 5;
  const __bf16* A   = probsb + (size_t)h * C_ * C_;               // [256 x 256]
  const __bf16* vtb = vt + (size_t)(h * R_ + r) * DK_ * C_;       // [64 x 256]
  v8f acc[4][2];
#pragma unroll
  for (int i = 0; i < 4; ++i)
#pragma unroll
    for (int j = 0; j < 2; ++j) acc[i][j] = (v8f){0.f,0.f,0.f,0.f,0.f,0.f,0.f,0.f};
  for (int k0 = 0; k0 < C_; k0 += 32) {
    v16bf a[4], b[2];
#pragma unroll
    for (int i = 0; i < 4; ++i) a[i] = load_a_rowmajor(A, C_, i0 + 16 * i, k0);
#pragma unroll
    for (int j = 0; j < 2; ++j) b[j] = load_b_wt(vtb, C_, d0 + 16 * j, k0);
#pragma unroll
    for (int i = 0; i < 4; ++i)
#pragma unroll
      for (int j = 0; j < 2; ++j) acc[i][j] = wmma_bf16(a[i], b[j], acc[i][j]);
  }
  __bf16* Cb = cb + (size_t)r * C_ * E_ + h * DK_;
#pragma unroll
  for (int i = 0; i < 4; ++i)
#pragma unroll
    for (int j = 0; j < 2; ++j)
      store_tile_bf16(Cb, E_, i0 + 16 * i, d0 + 16 * j, acc[i][j], nullptr, 1.0f);
}

// ---------------- launcher ----------------
extern "C" void kernel_launch(void* const* d_in, const int* in_sizes, int n_in,
                              void* d_out, int out_size, void* d_ws, size_t ws_size,
                              hipStream_t stream) {
  const float* x  = (const float*)d_in[0];
  const float* Wq = (const float*)d_in[1];
  const float* bq = (const float*)d_in[2];
  const float* Wk = (const float*)d_in[3];
  const float* bk = (const float*)d_in[4];
  const float* Wv = (const float*)d_in[5];
  const float* bv = (const float*)d_in[6];
  const float* Wo = (const float*)d_in[7];
  const float* bo = (const float*)d_in[8];

  float* out       = (float*)d_out;                       // [R,C,1,E] fp32
  float* probs_out = out + (size_t)M_ * E_;               // [H,1,C,C] fp32

  // workspace carve-up (all offsets 16B aligned)
  char* ws = (char*)d_ws;
  __bf16* xb  = (__bf16*)ws; ws += (size_t)M_ * E_ * sizeof(__bf16);  // reused as cb
  __bf16* qb  = (__bf16*)ws; ws += (size_t)M_ * E_ * sizeof(__bf16);
  __bf16* kb  = (__bf16*)ws; ws += (size_t)M_ * E_ * sizeof(__bf16);
  __bf16* vb  = (__bf16*)ws; ws += (size_t)M_ * E_ * sizeof(__bf16);
  __bf16* vt  = (__bf16*)ws; ws += (size_t)M_ * E_ * sizeof(__bf16);  // [h][r][d][j]
  __bf16* wqb = (__bf16*)ws; ws += (size_t)E_ * E_ * sizeof(__bf16);
  __bf16* wkb = (__bf16*)ws; ws += (size_t)E_ * E_ * sizeof(__bf16);
  __bf16* wvb = (__bf16*)ws; ws += (size_t)E_ * E_ * sizeof(__bf16);
  __bf16* wob = (__bf16*)ws; ws += (size_t)E_ * E_ * sizeof(__bf16);
  float*  attn   = (float*)ws;  ws += (size_t)H_ * C_ * C_ * sizeof(float);
  __bf16* probsb = (__bf16*)ws; ws += (size_t)H_ * C_ * C_ * sizeof(__bf16);
  __bf16* cb = xb;  // x no longer needed after the three projections

  const float q_scale = 0.125f / sqrtf((float)R_);  // DK^-0.5 / sqrt(R)

  // 1) convert inputs to bf16
  k_cvt_bf16<<<2048, 256, 0, stream>>>(x,  xb,  M_ * E_);
  k_cvt_bf16<<<576,  256, 0, stream>>>(Wq, wqb, E_ * E_);
  k_cvt_bf16<<<576,  256, 0, stream>>>(Wk, wkb, E_ * E_);
  k_cvt_bf16<<<576,  256, 0, stream>>>(Wv, wvb, E_ * E_);
  k_cvt_bf16<<<576,  256, 0, stream>>>(Wo, wob, E_ * E_);

  // 2) projections: q (scaled), k, v — 12288 waves each, 4 waves/block
  dim3 gemmGrid(3072), gemmBlk(128);
  k_proj_gemm<<<gemmGrid, gemmBlk, 0, stream>>>(xb, wqb, bq, qb, q_scale);
  k_proj_gemm<<<gemmGrid, gemmBlk, 0, stream>>>(xb, wkb, bk, kb, 1.0f);
  k_proj_gemm<<<gemmGrid, gemmBlk, 0, stream>>>(xb, wvb, bv, vb, 1.0f);

  // 3) per-head transpose of v for the context GEMM
  k_transpose_v<<<2048, 256, 0, stream>>>(vb, vt);

  // 4) pooled attention logits: 768 waves, 4 waves/block
  k_attn_gemm<<<192, 128, 0, stream>>>(qb, kb, attn);

  // 5) softmax (one block per (h,i) row)
  k_softmax<<<H_ * C_, 256, 0, stream>>>(attn, probs_out, probsb);

  // 6) context: 12288 waves
  k_ctx_gemm<<<gemmGrid, gemmBlk, 0, stream>>>(probsb, vt, cb);

  // 7) output projection -> d_out (fp32)
  k_out_gemm<<<gemmGrid, gemmBlk, 0, stream>>>(cb, wob, bo, out);
}
